// SingleHeadAttention_69088843923521
// MI455X (gfx1250) — matrix operand order repaired and checked by reading, not compiled, not run on hardware
//
#include <hip/hip_runtime.h>

typedef __attribute__((ext_vector_type(16))) __bf16 v16bf;
typedef __attribute__((ext_vector_type(8)))  __bf16 v8bf;
typedef __attribute__((ext_vector_type(8)))  float  v8f;

#define WMMA_BF16(a, b, c) \
    __builtin_amdgcn_wmma_f32_16x16x32_bf16(false, (a), false, (b), (short)0, (c), false, false)

// ---------------------------------------------------------------------------
// Kernel 1: projections  O = X * W   (X: [16384,1024] f32, W: [1024,128] f32)
// One block = 128 threads = 4 waves; each wave -> 16 rows x 128 cols.
// blockIdx.y selects which projection (q / k / v). q is pre-scaled by 1/sqrt(dk).
// ---------------------------------------------------------------------------
__global__ __launch_bounds__(128)
void sha_proj_kernel(const float* __restrict__ Xq, const float* __restrict__ Xk,
                     const float* __restrict__ Xv,
                     const float* __restrict__ WQ, const float* __restrict__ WK,
                     const float* __restrict__ WV,
                     __bf16* __restrict__ Oq, __bf16* __restrict__ Ok,
                     __bf16* __restrict__ Ov)
{
    const int which = blockIdx.y;
    const float* X = (which == 0) ? Xq : (which == 1) ? Xk : Xv;
    const float* W = (which == 0) ? WQ : (which == 1) ? WK : WV;
    __bf16*      O = (which == 0) ? Oq : (which == 1) ? Ok : Ov;
    const float  oscale = (which == 0) ? 0.08838834764831845f : 1.0f;  // 1/sqrt(128)

    const int t    = threadIdx.x;
    const int wave = t >> 5;
    const int lane = t & 31;
    const int l16  = lane & 15;
    const int hi   = lane >> 4;                 // 0 or 1 (half-wave)
    const int mrow = blockIdx.x * 64 + wave * 16 + l16;   // A-frag row

    __shared__ __align__(32) __bf16 WT[128][32];   // transposed W chunk: WT[n][k]

    const v8f vzero = {0.f,0.f,0.f,0.f,0.f,0.f,0.f,0.f};
    v8f acc[8];
#pragma unroll
    for (int n = 0; n < 8; ++n) acc[n] = vzero;

    for (int kb = 0; kb < 1024; kb += 32) {
        __syncthreads();
        {   // cooperative load of W[kb..kb+31][0..127], stored transposed
            const int krow = t >> 2;            // 0..31
            const int cb   = (t & 3) * 32;      // 32 contiguous cols
            const float* wrow = W + (kb + krow) * 128 + cb;
#pragma unroll
            for (int c = 0; c < 32; ++c)
                WT[cb + c][krow] = (__bf16)wrow[c];
        }
        __syncthreads();

        // A fragment: row mrow, k = kb + e + 16*hi
        const float* xr = X + (size_t)mrow * 1024 + kb;
        union { v16bf v; __bf16 h[16]; } a;
#pragma unroll
        for (int e = 0; e < 8; ++e) a.h[e]     = (__bf16)xr[hi * 16 + e];
#pragma unroll
        for (int e = 0; e < 8; ++e) a.h[8 + e] = (__bf16)xr[hi * 16 + 8 + e];

#pragma unroll
        for (int n = 0; n < 8; ++n) {
            const v16bf b = *(const v16bf*)&WT[n * 16 + l16][hi * 16];
            acc[n] = WMMA_BF16(a.v, b, acc[n]);
        }
    }

    // C layout: row = base + r + 8*hi, col = n*16 + l16
    const int mbase = blockIdx.x * 64 + wave * 16 + hi * 8;
#pragma unroll
    for (int n = 0; n < 8; ++n)
#pragma unroll
        for (int r = 0; r < 8; ++r)
            O[(size_t)(mbase + r) * 128 + n * 16 + l16] = (__bf16)(acc[n][r] * oscale);
}

// ---------------------------------------------------------------------------
// Kernel 2: causal flash attention over bf16 q/k/v, f32 output.
// Block = 128 threads (4 waves); block owns 64 query rows of one batch.
// K tile staged via GLOBAL_LOAD_ASYNC_TO_LDS_B128 (ASYNCcnt path).
// ---------------------------------------------------------------------------
__global__ __launch_bounds__(128)
void sha_attn_kernel(const __bf16* __restrict__ Q, const __bf16* __restrict__ K,
                     const __bf16* __restrict__ V, float* __restrict__ Out)
{
    constexpr int S  = 4096;
    constexpr int DK = 128;

    const int b    = blockIdx.y;
    const int t    = threadIdx.x;
    const int wave = t >> 5;
    const int lane = t & 31;
    const int l16  = lane & 15;
    const int hi   = lane >> 4;
    const int q0   = blockIdx.x * 64;            // block's query base
    const int mw   = q0 + wave * 16;             // wave's 16-row strip

    const __bf16* Qb = Q + (size_t)b * S * DK;
    const __bf16* Kb = K + (size_t)b * S * DK;
    const __bf16* Vb = V + (size_t)b * S * DK;

    __shared__ __align__(32) __bf16 Kt[32][128];     // K tile, row-major [j][dk]
    __shared__ __align__(32) __bf16 VT[128][32];     // V tile transposed [n][j]
    __shared__ __align__(32) __bf16 Pt[4][16][32];   // per-wave P re-layout tile

    // Preload 4 q A-fragments (DK = 4 chunks of K=32); q already pre-scaled.
    v16bf qa[4];
    {
        const __bf16* qr = Qb + (size_t)(mw + l16) * DK;
#pragma unroll
        for (int c = 0; c < 4; ++c) {
            v8bf lo = *(const v8bf*)&qr[c * 32 + hi * 16];
            v8bf hh = *(const v8bf*)&qr[c * 32 + hi * 16 + 8];
            qa[c] = __builtin_shufflevector(lo, hh,
                     0,1,2,3,4,5,6,7,8,9,10,11,12,13,14,15);
        }
    }

    const v8f vzero = {0.f,0.f,0.f,0.f,0.f,0.f,0.f,0.f};
    v8f acc[8];
#pragma unroll
    for (int n = 0; n < 8; ++n) acc[n] = vzero;
    float mrun[8], lrun[8];
#pragma unroll
    for (int r = 0; r < 8; ++r) { mrun[r] = -1e30f; lrun[r] = 0.0f; }

    const int jend = q0 + 64;                    // causal: j <= q0+63

    // Per-thread staging coordinates (thread -> one K row chunk + one V row chunk)
    const int jr = t >> 2;                       // 0..31
    const int cb = (t & 3) * 32;                 // 32 contiguous cols
    const unsigned ldsK = (unsigned)(size_t)&Kt[jr][cb];   // LDS byte offset

    for (int jb = 0; jb < jend; jb += 32) {
        __syncthreads();
        {   // stage K asynchronously (global -> LDS direct, ASYNCcnt), V transposed
            const __bf16* krow = Kb + (size_t)(jb + jr) * DK + cb;
            const __bf16* vrow = Vb + (size_t)(jb + jr) * DK + cb;
            if (jb + 32 < jend)                  // prefetch next K tile
                __builtin_prefetch(krow + 32 * DK, 0, 3);
            const unsigned long long gK = (unsigned long long)(size_t)krow;
            // offset: immediate applies to BOTH the global and LDS addresses
            asm volatile(
                "global_load_async_to_lds_b128 %0, %1, off\n\t"
                "global_load_async_to_lds_b128 %0, %1, off offset:16\n\t"
                "global_load_async_to_lds_b128 %0, %1, off offset:32\n\t"
                "global_load_async_to_lds_b128 %0, %1, off offset:48"
                :: "v"(ldsK), "v"(gK) : "memory");
#pragma unroll
            for (int c = 0; c < 32; ++c) VT[cb + c][jr] = vrow[c];
            asm volatile("s_wait_asynccnt 0" ::: "memory");
        }
        __syncthreads();

        // scores: two 16x16 tiles (j = jb..jb+15, jb+16..jb+31)
        v8f s0 = vzero, s1 = vzero;
#pragma unroll
        for (int c = 0; c < 4; ++c) {
            v16bf b0 = *(const v16bf*)&Kt[l16][c * 32 + hi * 16];
            v16bf b1 = *(const v16bf*)&Kt[16 + l16][c * 32 + hi * 16];
            s0 = WMMA_BF16(qa[c], b0, s0);
            s1 = WMMA_BF16(qa[c], b1, s1);
        }

        // online softmax (C layout: lane holds rows mg..mg+7, one column each)
        const int mg = mw + hi * 8;
        const int j0 = jb + l16, j1 = jb + 16 + l16;
        float p0[8], p1[8];
#pragma unroll
        for (int r = 0; r < 8; ++r) {
            const int row = mg + r;
            float a0 = (j0 <= row) ? s0[r] : -1e30f;
            float a1 = (j1 <= row) ? s1[r] : -1e30f;
            float tmax = fmaxf(a0, a1);
            tmax = fmaxf(tmax, __shfl_xor(tmax, 1));
            tmax = fmaxf(tmax, __shfl_xor(tmax, 2));
            tmax = fmaxf(tmax, __shfl_xor(tmax, 4));
            tmax = fmaxf(tmax, __shfl_xor(tmax, 8));
            const float mnew = fmaxf(mrun[r], tmax);
            const float corr = __expf(mrun[r] - mnew);
            mrun[r] = mnew;
            const float e0 = __expf(a0 - mnew);
            const float e1 = __expf(a1 - mnew);
            float ps = e0 + e1;
            ps += __shfl_xor(ps, 1);
            ps += __shfl_xor(ps, 2);
            ps += __shfl_xor(ps, 4);
            ps += __shfl_xor(ps, 8);
            lrun[r] = lrun[r] * corr + ps;
            p0[r] = e0; p1[r] = e1;
#pragma unroll
            for (int n = 0; n < 8; ++n) acc[n][r] *= corr;
        }

        // re-layout P (C layout -> A fragment) through per-wave LDS tile
#pragma unroll
        for (int r = 0; r < 8; ++r) {
            Pt[wave][hi * 8 + r][l16]      = (__bf16)p0[r];
            Pt[wave][hi * 8 + r][16 + l16] = (__bf16)p1[r];
        }
        asm volatile("s_wait_dscnt 0" ::: "memory");   // intra-wave LDS RAW
        v8bf plo = *(const v8bf*)&Pt[wave][l16][hi * 16];
        v8bf phh = *(const v8bf*)&Pt[wave][l16][hi * 16 + 8];
        v16bf pa = __builtin_shufflevector(plo, phh,
                    0,1,2,3,4,5,6,7,8,9,10,11,12,13,14,15);

        // out += P @ V   (8 n-tiles; B frags are contiguous reads of VT)
#pragma unroll
        for (int n = 0; n < 8; ++n) {
            v16bf bv = *(const v16bf*)&VT[n * 16 + l16][hi * 16];
            acc[n] = WMMA_BF16(pa, bv, acc[n]);
        }
    }

    // epilogue: divide by row sums, write f32 output
#pragma unroll
    for (int n = 0; n < 8; ++n)
#pragma unroll
        for (int r = 0; r < 8; ++r)
            Out[((size_t)b * S + mw + hi * 8 + r) * DK + n * 16 + l16] =
                acc[n][r] / lrun[r];
}

// ---------------------------------------------------------------------------
extern "C" void kernel_launch(void* const* d_in, const int* in_sizes, int n_in,
                              void* d_out, int out_size, void* d_ws, size_t ws_size,
                              hipStream_t stream)
{
    (void)in_sizes; (void)n_in; (void)out_size; (void)ws_size;
    const float* Xq = (const float*)d_in[0];
    const float* Xk = (const float*)d_in[1];
    const float* Xv = (const float*)d_in[2];
    const float* WQ = (const float*)d_in[3];
    const float* WK = (const float*)d_in[4];
    const float* WV = (const float*)d_in[5];

    constexpr size_t M  = 4 * 4096;   // B*S
    constexpr size_t DK = 128;
    __bf16* qws = (__bf16*)d_ws;
    __bf16* kws = qws + M * DK;
    __bf16* vws = kws + M * DK;

    dim3 gProj((unsigned)(M / 64), 3, 1);
    sha_proj_kernel<<<gProj, dim3(128), 0, stream>>>(Xq, Xk, Xv, WQ, WK, WV,
                                                     qws, kws, vws);

    dim3 gAttn(4096 / 64, 4, 1);
    sha_attn_kernel<<<gAttn, dim3(128), 0, stream>>>(qws, kws, vws, (float*)d_out);
}
